// TensorBase_59450937311942
// MI455X (gfx1250) — matrix operand order repaired and checked by reading, not compiled, not run on hardware
//
#include <hip/hip_runtime.h>

typedef __attribute__((ext_vector_type(16))) _Float16 v16h;
typedef __attribute__((ext_vector_type(8)))  _Float16 v8h;
typedef __attribute__((ext_vector_type(8)))  float    v8f;

#define GRID_N        128
#define NRAYS         4096
#define NSAMP         128
#define TILES_PER_RAY (NSAMP / 16)                 // 8
#define NTILES        (NRAYS * NSAMP / 16)         // 32768

#define STEP_SZ     (6.0f / 127.0f)                // mean(3/127)*2
#define DIST_SCALE  25.0f
#define DEN_SHIFT   (-10.0f)
#define W_THRES     1e-4f

#define K1PAD 416   // 390 padded to 13*32
#define KT1   13
#define NT1   8     // 128 outputs / 16
#define KT2   4     // 128 / 32
#define NT2   8
#define KT3   4     // 128 / 32 (output padded 3 -> 16)

// ---------- helpers ----------
__device__ __forceinline__ float fetch3(const float* v, int z, int y, int x) {
  return v[((size_t)z * GRID_N + y) * GRID_N + x];
}

__device__ __forceinline__ float trilerp(const float* v, int x0, int x1, int y0, int y1,
                                         int z0, int z1, float fx, float fy, float fz) {
  float c00 = fetch3(v, z0, y0, x0) * (1.f - fx) + fetch3(v, z0, y0, x1) * fx;
  float c01 = fetch3(v, z0, y1, x0) * (1.f - fx) + fetch3(v, z0, y1, x1) * fx;
  float c10 = fetch3(v, z1, y0, x0) * (1.f - fx) + fetch3(v, z1, y0, x1) * fx;
  float c11 = fetch3(v, z1, y1, x0) * (1.f - fx) + fetch3(v, z1, y1, x1) * fx;
  float c0 = c00 * (1.f - fy) + c01 * fy;
  float c1 = c10 * (1.f - fy) + c11 * fy;
  return c0 * (1.f - fz) + c1 * fz;
}

__device__ __forceinline__ void gcoord(float g, int& i0, int& i1, float& f) {
  int t = (int)floorf(g);
  i0 = t < 0 ? 0 : (t > GRID_N - 1 ? GRID_N - 1 : t);
  i1 = (i0 + 1 > GRID_N - 1) ? GRID_N - 1 : i0 + 1;
  f = g - (float)i0;
  f = f < 0.f ? 0.f : (f > 1.f ? 1.f : f);
}

// ---------- kernel 0: pack MLP weights into WMMA B-fragment layout (f16) ----------
// B fragment (32x16 f16): lane L -> column N = L&15, half h -> K = (L>>4)*8 + h + (h >= 8 ? 8 : 0).
// Tile stored as 512 contiguous halves: [lane][h].
__global__ void pack_weights_kernel(const float* __restrict__ W1, const float* __restrict__ W2,
                                    const float* __restrict__ W3,
                                    _Float16* __restrict__ W1p, _Float16* __restrict__ W2p,
                                    _Float16* __restrict__ W3p) {
  int idx = blockIdx.x * blockDim.x + threadIdx.x;
  const int NW1 = NT1 * KT1 * 512;  // 53248
  const int NW2 = NT2 * KT2 * 512;  // 16384
  const int NW3 = KT3 * 512;        // 2048
  if (idx >= NW1 + NW2 + NW3) return;
  if (idx < NW1) {
    int h = idx & 15, lane = (idx >> 4) & 31, t = idx >> 9;
    int kt = t % KT1, nt = t / KT1;
    int col = lane & 15, hi = lane >> 4;
    int k = kt * 32 + (h < 8 ? hi * 8 + h : 16 + hi * 8 + (h - 8)), n = nt * 16 + col;
    W1p[idx] = (k < 390) ? (_Float16)W1[k * 128 + n] : (_Float16)0.f;
  } else if (idx < NW1 + NW2) {
    int i = idx - NW1;
    int h = i & 15, lane = (i >> 4) & 31, t = i >> 9;
    int kt = t & 3, nt = t >> 2;
    int col = lane & 15, hi = lane >> 4;
    int k = kt * 32 + (h < 8 ? hi * 8 + h : 16 + hi * 8 + (h - 8)), n = nt * 16 + col;
    W2p[i] = (_Float16)W2[k * 128 + n];
  } else {
    int i = idx - NW1 - NW2;
    int h = i & 15, lane = (i >> 4) & 31, kt = i >> 9;
    int col = lane & 15, hi = lane >> 4;
    int k = kt * 32 + (h < 8 ? hi * 8 + h : 16 + hi * 8 + (h - 8));
    W3p[i] = (col < 3) ? (_Float16)W3[k * 3 + col] : (_Float16)0.f;
  }
}

// ---------- kernel A: ray march, density, transmittance (thread per ray) ----------
__global__ void ray_march_kernel(const float* __restrict__ rays_o, const float* __restrict__ rays_d,
                                 const float* __restrict__ dvol,
                                 float* __restrict__ weight, float* __restrict__ tmin_out,
                                 float* __restrict__ acc_out, float* __restrict__ depth_out,
                                 float* __restrict__ rgb_part) {
  int ray = blockIdx.x * blockDim.x + threadIdx.x;
  if (ray >= NRAYS) return;
  float ox = rays_o[ray * 3 + 0], oy = rays_o[ray * 3 + 1], oz = rays_o[ray * 3 + 2];
  float dx = rays_d[ray * 3 + 0], dy = rays_d[ray * 3 + 1], dz = rays_d[ray * 3 + 2];
  float vx = (dx == 0.f) ? 1e-6f : dx;
  float vy = (dy == 0.f) ? 1e-6f : dy;
  float vz = (dz == 0.f) ? 1e-6f : dz;
  float tx = fminf((1.5f - ox) / vx, (-1.5f - ox) / vx);
  float ty = fminf((1.5f - oy) / vy, (-1.5f - oy) / vy);
  float tz = fminf((1.5f - oz) / vz, (-1.5f - oz) / vz);
  float tmin = fmaxf(fmaxf(tx, ty), tz);
  tmin = fminf(fmaxf(tmin, 2.0f), 6.0f);
  tmin_out[ray] = tmin;

  // zero partial rgb slots (d_ws is poisoned; skipped tiles never write)
  #pragma unroll
  for (int t = 0; t < TILES_PER_RAY * 3; t++) rgb_part[ray * TILES_PER_RAY * 3 + t] = 0.f;

  float T = 1.f, acc = 0.f, depth = 0.f;
  for (int s = 0; s < NSAMP; s++) {
    float z = tmin + STEP_SZ * (float)s;
    float px = ox + dx * z, py = oy + dy * z, pz = oz + dz * z;
    bool valid = (px >= -1.5f) && (px <= 1.5f) && (py >= -1.5f) && (py <= 1.5f) &&
                 (pz >= -1.5f) && (pz <= 1.5f);
    float sigma = 0.f;
    if (valid) {
      int x0, x1, y0, y1, z0, z1;
      float fx, fy, fz;
      gcoord((px + 1.5f) * (127.0f / 3.0f), x0, x1, fx);
      gcoord((py + 1.5f) * (127.0f / 3.0f), y0, y1, fy);
      gcoord((pz + 1.5f) * (127.0f / 3.0f), z0, z1, fz);
      float g = trilerp(dvol, x0, x1, y0, y1, z0, z1, fx, fy, fz);
      float u = g + DEN_SHIFT;
      sigma = (u > 20.f) ? u : log1pf(expf(u));  // softplus
    }
    float dist = (s == NSAMP - 1) ? 0.f : STEP_SZ;
    float alpha = 1.f - expf(-sigma * dist * DIST_SCALE);
    float w = alpha * T;
    T *= (1.f - alpha + 1e-10f);
    weight[ray * NSAMP + s] = w;
    acc += w;
    depth += w * z;
  }
  acc_out[ray] = acc;
  depth_out[ray] = depth;
}

// ---------- kernel B: appearance gather + 3-layer MLP via WMMA (one wave per 16-sample tile) ----------
__global__ __launch_bounds__(32)
void shade_kernel(const float* __restrict__ rays_o, const float* __restrict__ rays_d,
                  const float* __restrict__ app_vol,
                  const float* __restrict__ tmin_arr, const float* __restrict__ weight,
                  const _Float16* __restrict__ W1p, const _Float16* __restrict__ W2p,
                  const _Float16* __restrict__ W3p,
                  const float* __restrict__ b1, const float* __restrict__ b2,
                  const float* __restrict__ b3,
                  float* __restrict__ rgb_part) {
  __shared__ __align__(32) _Float16 Ain[16][K1PAD];  // mlp_in tile, 16 x 416
  __shared__ __align__(32) _Float16 H1[16][128];
  __shared__ __align__(32) _Float16 H2[16][128];
  __shared__ float wtile[16];
  __shared__ float sg[16][3];
  __shared__ float ppart[2][3];

  const int lane = threadIdx.x;
  const int tile = blockIdx.x;
  const int base = tile * 16;
  const int ray = base >> 7;    // 128 samples per ray, tiles align inside a ray
  const int srow = base & 127;

  if (lane < 16) wtile[lane] = weight[base + lane];
  __syncthreads();
  bool anyv = false;
  #pragma unroll
  for (int i = 0; i < 16; i++) anyv = anyv || (wtile[i] > W_THRES);
  if (!anyv) return;  // uniform exit: EXEC stays all-ones for WMMA below

  const float ox = rays_o[ray * 3 + 0], oy = rays_o[ray * 3 + 1], oz = rays_o[ray * 3 + 2];
  const float dx = rays_d[ray * 3 + 0], dy = rays_d[ray * 3 + 1], dz = rays_d[ray * 3 + 2];
  const float tmin = tmin_arr[ray];

  if (lane < 16) {
    float z = tmin + STEP_SZ * (float)(srow + lane);
    sg[lane][0] = (ox + dx * z + 1.5f) * (127.0f / 3.0f);
    sg[lane][1] = (oy + dy * z + 1.5f) * (127.0f / 3.0f);
    sg[lane][2] = (oz + dz * z + 1.5f) * (127.0f / 3.0f);
  }
  // zero K padding columns 390..415
  for (int j = lane; j < 16 * (K1PAD - 390); j += 32) {
    int r = j / (K1PAD - 390), c2 = 390 + j % (K1PAD - 390);
    Ain[r][c2] = (_Float16)0.f;
  }
  __syncthreads();

  // features: lane c (0..26) = app channel, lanes 27..29 = view dir components
  const int c = lane;
  for (int m = 0; m < 16; m++) {
    if (c < 27) {
      int x0, x1, y0, y1, z0, z1;
      float fx, fy, fz;
      gcoord(sg[m][0], x0, x1, fx);
      gcoord(sg[m][1], y0, y1, fy);
      gcoord(sg[m][2], z0, z1, fz);
      const float* vc = app_vol + (size_t)c * (GRID_N * GRID_N * GRID_N);
      float feat = trilerp(vc, x0, x1, y0, y1, z0, z1, fx, fy, fz);
      Ain[m][c] = (_Float16)feat;
      #pragma unroll
      for (int f = 0; f < 6; f++) {
        float p = feat * (float)(1 << f);
        Ain[m][30 + c * 6 + f] = (_Float16)__sinf(p);
        Ain[m][192 + c * 6 + f] = (_Float16)__cosf(p);
      }
    } else if (c < 30) {
      int j = c - 27;
      float v = (j == 0) ? dx : ((j == 1) ? dy : dz);
      Ain[m][27 + j] = (_Float16)v;
      #pragma unroll
      for (int f = 0; f < 6; f++) {
        float p = v * (float)(1 << f);
        Ain[m][354 + j * 6 + f] = (_Float16)__sinf(p);
        Ain[m][372 + j * 6 + f] = (_Float16)__cosf(p);
      }
    }
  }
  __syncthreads();

  const int col = lane & 15;  // = A-fragment row and B/C column
  const int hi = lane >> 4;

  // A fragments, layer 1: per ISA layout two contiguous 8-half chunks per lane
  v16h a1[KT1];
  #pragma unroll
  for (int kt = 0; kt < KT1; kt++) {
    int kb = kt * 32 + hi * 8;
    v8h l8 = *(const v8h*)&Ain[col][kb];
    v8h h8 = *(const v8h*)&Ain[col][kb + 16];
    a1[kt] = __builtin_shufflevector(l8, h8, 0, 1, 2, 3, 4, 5, 6, 7, 8, 9, 10, 11, 12, 13, 14, 15);
  }
  // ---- layer 1: 16x416 @ 416x128 ----
  #pragma unroll
  for (int nt = 0; nt < NT1; nt++) {
    v8f acc = {};
    #pragma unroll
    for (int kt = 0; kt < KT1; kt++) {
      v16h b = *(const v16h*)(W1p + (size_t)((nt * KT1 + kt) * 512 + lane * 16));
      acc = __builtin_amdgcn_wmma_f32_16x16x32_f16(false, a1[kt], false, b, (short)0, acc,
                                                   false, false);
    }
    float bias = b1[nt * 16 + col];
    #pragma unroll
    for (int r = 0; r < 8; r++) {
      float v = acc[r] + bias;
      H1[r + 8 * hi][nt * 16 + col] = (_Float16)(v > 0.f ? v : 0.f);
    }
  }
  __syncthreads();

  // ---- layer 2: 16x128 @ 128x128 ----
  v16h a2[KT2];
  #pragma unroll
  for (int kt = 0; kt < KT2; kt++) {
    int kb = kt * 32 + hi * 8;
    v8h l8 = *(const v8h*)&H1[col][kb];
    v8h h8 = *(const v8h*)&H1[col][kb + 16];
    a2[kt] = __builtin_shufflevector(l8, h8, 0, 1, 2, 3, 4, 5, 6, 7, 8, 9, 10, 11, 12, 13, 14, 15);
  }
  #pragma unroll
  for (int nt = 0; nt < NT2; nt++) {
    v8f acc = {};
    #pragma unroll
    for (int kt = 0; kt < KT2; kt++) {
      v16h b = *(const v16h*)(W2p + (size_t)((nt * KT2 + kt) * 512 + lane * 16));
      acc = __builtin_amdgcn_wmma_f32_16x16x32_f16(false, a2[kt], false, b, (short)0, acc,
                                                   false, false);
    }
    float bias = b2[nt * 16 + col];
    #pragma unroll
    for (int r = 0; r < 8; r++) {
      float v = acc[r] + bias;
      H2[r + 8 * hi][nt * 16 + col] = (_Float16)(v > 0.f ? v : 0.f);
    }
  }
  __syncthreads();

  // ---- layer 3: 16x128 @ 128x16 (cols 0..2 valid) ----
  v8f acc3 = {};
  #pragma unroll
  for (int kt = 0; kt < KT3; kt++) {
    int kb = kt * 32 + hi * 8;
    v8h l8 = *(const v8h*)&H2[col][kb];
    v8h h8 = *(const v8h*)&H2[col][kb + 16];
    v16h a3 = __builtin_shufflevector(l8, h8, 0, 1, 2, 3, 4, 5, 6, 7, 8, 9, 10, 11, 12, 13, 14, 15);
    v16h b = *(const v16h*)(W3p + (size_t)(kt * 512 + lane * 16));
    acc3 = __builtin_amdgcn_wmma_f32_16x16x32_f16(false, a3, false, b, (short)0, acc3,
                                                  false, false);
  }

  // composite this tile's weighted rgb (deterministic: per-(ray,tile) slot, no atomics)
  if (col < 3) {
    float bias = b3[col];
    float s = 0.f;
    #pragma unroll
    for (int r = 0; r < 8; r++) {
      int m = r + 8 * hi;  // C/D layout: lane holds rows r + 8*hi
      float w = wtile[m];
      if (w > W_THRES) {
        float x = acc3[r] + bias;
        s += w * (1.0f / (1.0f + expf(-x)));  // sigmoid
      }
    }
    ppart[hi][col] = s;
  }
  __syncthreads();
  if (hi == 0 && col < 3) {
    rgb_part[(ray * TILES_PER_RAY + (srow >> 4)) * 3 + col] = ppart[0][col] + ppart[1][col];
  }
}

// ---------- kernel C: finalize per-ray output ----------
__global__ void finalize_kernel(const float* __restrict__ rgb_part, const float* __restrict__ acc,
                                const float* __restrict__ depth, float* __restrict__ out) {
  int ray = blockIdx.x * blockDim.x + threadIdx.x;
  if (ray >= NRAYS) return;
  float bg = 1.f - acc[ray];
  #pragma unroll
  for (int cdx = 0; cdx < 3; cdx++) {
    float s = 0.f;
    for (int t = 0; t < TILES_PER_RAY; t++) s += rgb_part[(ray * TILES_PER_RAY + t) * 3 + cdx];
    float v = s + bg;
    out[ray * 4 + cdx] = v < 0.f ? 0.f : (v > 1.f ? 1.f : v);
  }
  out[ray * 4 + 3] = depth[ray];
}

extern "C" void kernel_launch(void* const* d_in, const int* in_sizes, int n_in,
                              void* d_out, int out_size, void* d_ws, size_t ws_size,
                              hipStream_t stream) {
  (void)in_sizes; (void)n_in; (void)out_size; (void)ws_size;
  const float* rays_o = (const float*)d_in[0];
  const float* rays_d = (const float*)d_in[1];
  const float* dvol   = (const float*)d_in[2];
  const float* avol   = (const float*)d_in[3];
  const float* W1     = (const float*)d_in[4];
  const float* b1     = (const float*)d_in[5];
  const float* W2     = (const float*)d_in[6];
  const float* b2     = (const float*)d_in[7];
  const float* W3     = (const float*)d_in[8];
  const float* b3     = (const float*)d_in[9];
  // d_in[10] = n_samples (fixed 128 per reference setup)

  char* ws = (char*)d_ws;
  float* weight = (float*)ws;     ws += (size_t)NRAYS * NSAMP * 4;          // 2 MB
  _Float16* W1p = (_Float16*)ws;  ws += (size_t)NT1 * KT1 * 512 * 2;        // 104 KB
  _Float16* W2p = (_Float16*)ws;  ws += (size_t)NT2 * KT2 * 512 * 2;        // 32 KB
  _Float16* W3p = (_Float16*)ws;  ws += (size_t)KT3 * 512 * 2;              // 4 KB
  float* tminb  = (float*)ws;     ws += (size_t)NRAYS * 4;
  float* accb   = (float*)ws;     ws += (size_t)NRAYS * 4;
  float* depthb = (float*)ws;     ws += (size_t)NRAYS * 4;
  float* rgbp   = (float*)ws;     ws += (size_t)NRAYS * TILES_PER_RAY * 3 * 4;

  const int packN = (NT1 * KT1 + NT2 * KT2 + KT3) * 512;
  pack_weights_kernel<<<(packN + 255) / 256, 256, 0, stream>>>(W1, W2, W3, W1p, W2p, W3p);
  ray_march_kernel<<<NRAYS / 256, 256, 0, stream>>>(rays_o, rays_d, dvol, weight, tminb, accb,
                                                    depthb, rgbp);
  shade_kernel<<<NTILES, 32, 0, stream>>>(rays_o, rays_d, avol, tminb, weight, W1p, W2p, W3p,
                                          b1, b2, b3, rgbp);
  finalize_kernel<<<NRAYS / 256, 256, 0, stream>>>(rgbp, accb, depthb, (float*)d_out);
}
